// DecisionMetaMamba_17549236371942
// MI455X (gfx1250) — compile-verified, matrix operand order checked by reading, not run on hardware
//
#include <hip/hip_runtime.h>
#include <hip/hip_bf16.h>
#include <math.h>

// ---------------------------------------------------------------------------
// Model dims
// ---------------------------------------------------------------------------
#define DM    512      // D_MODEL
#define NL    6        // N_LAYER
#define DI    1024     // D_INNER
#define DS    16       // D_STATE
#define DC    4        // D_CONV
#define DTR   32       // DT_RANK
#define SDIM  17
#define ADIM  6
#define BATCH 16
#define SEQ   256
#define L3    (3*SEQ)          // 768 stacked tokens per batch element
#define ROWS  (BATCH*L3)       // 12288 token rows

typedef __bf16 bf16;
typedef bf16  v16bf __attribute__((ext_vector_type(16)));
typedef float v8f   __attribute__((ext_vector_type(8)));

__device__ __forceinline__ bf16 f2bf(float f) {
    union { float f; unsigned int u; } v; v.f = f;
    unsigned int r = v.u + 0x7FFFu + ((v.u >> 16) & 1u);   // round-to-nearest-even
    unsigned short h = (unsigned short)(r >> 16);
    return __builtin_bit_cast(bf16, h);
}
__device__ __forceinline__ float silu_f(float x) { return x / (1.0f + __expf(-x)); }

// ---------------------------------------------------------------------------
// Weight pre-pack: fp32 [K,N] -> bf16 WMMA B fragments.
// Fragment (ntile, ktile) holds 32 lanes x 16 bf16, laid out exactly as the
// v_wmma_f32_16x16x32_bf16 B operand wants it:
//   lane: col = ntile*16 + (lane&15), kb = ktile*32 + (lane<16 ? 0 : 8)
//   half i   (i<8)  = B[kb+i][col]
//   half 8+i (i<8)  = B[kb+16+i][col]
// Storage: P[(ntile*(K/32) + ktile)*512 + lane*16 + i]  (32B/lane, aligned)
// ---------------------------------------------------------------------------
__global__ __launch_bounds__(128)
void pack_b_kernel(const float* __restrict__ B, bf16* __restrict__ P, int K, int N)
{
    const int frag = blockIdx.x * 4 + (threadIdx.x >> 5);
    const int lane = threadIdx.x & 31;
    const int nfragK = K >> 5;
    if (frag >= nfragK * (N >> 4)) return;
    const int ntile = frag / nfragK;
    const int ktile = frag % nfragK;
    const int nc = ntile * 16 + (lane & 15);
    const int kb = ktile * 32 + ((lane < 16) ? 0 : 8);
    bf16* dst = P + (size_t)frag * 512 + lane * 16;
#pragma unroll
    for (int i = 0; i < 8; ++i) {
        dst[i]     = f2bf(B[(size_t)(kb + i) * N + nc]);
        dst[8 + i] = f2bf(B[(size_t)(kb + 16 + i) * N + nc]);
    }
}

// ---------------------------------------------------------------------------
// WMMA bf16 GEMM: C = act(A[M,K](bf16, row-major) * Bp(packed) + bias)(+res)
// One wave computes a (16*MT)x(16*NT) tile: MT A fragments and NT B fragments
// per K-step feed MT*NT v_wmma ops. 4 waves per block span 64*NT columns.
// ---------------------------------------------------------------------------
#define ACT_NONE     0
#define ACT_SOFTPLUS 1

template<int MT, int NT, int ACT, bool HAS_BIAS, bool HAS_RES, bool BF_OUT>
__global__ __launch_bounds__(128)
void gemm_wmma_bf16(const bf16* __restrict__ A, int lda,
                    const bf16* __restrict__ Bp,
                    const float* __restrict__ bias,
                    const float* __restrict__ res,
                    float* __restrict__ C,
                    bf16*  __restrict__ Cbf,
                    int M, int N, int K)
{
    const int lane = threadIdx.x & 31;
    const int wave = threadIdx.x >> 5;
    const int tm   = blockIdx.y * (16 * MT);               // M tile base
    const int tn0  = (blockIdx.x * 4 + wave) * (16 * NT);  // first N tile of strip
    if (tn0 >= N || tm >= M) return;

    const int nfragK = K >> 5;
    const int kb = (lane < 16) ? 0 : 8;                    // half-wave K base

    v8f acc[MT][NT];
#pragma unroll
    for (int mt = 0; mt < MT; ++mt)
#pragma unroll
        for (int j = 0; j < NT; ++j) acc[mt][j] = (v8f){};

    for (int k0 = 0; k0 < K; k0 += 32) {
        // ---- MT A fragments: two contiguous 16B loads each ----
        v16bf a[MT];
#pragma unroll
        for (int mt = 0; mt < MT; ++mt) {
            const int mr = tm + mt * 16 + (lane & 15);
            const bf16* ap = A + (size_t)mr * lda + (k0 + kb);
            __builtin_prefetch(ap + 32, 0, 1);             // speculative; OOB is dropped
#pragma unroll
            for (int i = 0; i < 8; ++i) { a[mt][i] = ap[i]; a[mt][8 + i] = ap[16 + i]; }
        }

        // ---- NT packed-B fragments, each reused by MT wmma ops ----
#pragma unroll
        for (int j = 0; j < NT; ++j) {
            const bf16* bp = Bp + ((size_t)((tn0 >> 4) + j) * nfragK + (k0 >> 5)) * 512
                                + lane * 16;
            __builtin_prefetch(bp + 512, 0, 1);            // next K fragment
            const v16bf b = *(const v16bf*)bp;
#pragma unroll
            for (int mt = 0; mt < MT; ++mt)
                acc[mt][j] = __builtin_amdgcn_wmma_f32_16x16x32_bf16(
                                 false, a[mt], false, b, (short)0, acc[mt][j],
                                 false, false);
        }
    }

#pragma unroll
    for (int mt = 0; mt < MT; ++mt) {
        const int mbase = tm + mt * 16 + ((lane >= 16) ? 8 : 0);
#pragma unroll
        for (int j = 0; j < NT; ++j) {
            const int nc = tn0 + j * 16 + (lane & 15);
            float bv;
            if (HAS_BIAS) bv = bias[nc];
#pragma unroll
            for (int r = 0; r < 8; ++r) {
                const int m = mbase + r;
                float v = acc[mt][j][r];
                if (HAS_BIAS) v += bv;                      // no +0.0 in no-bias path
                if (ACT == ACT_SOFTPLUS) v = (v > 20.0f) ? v : log1pf(__expf(v));
                if (HAS_RES) v += res[(size_t)m * N + nc];
                C[(size_t)m * N + nc] = v;
                if (BF_OUT) Cbf[(size_t)m * N + nc] = f2bf(v);
            }
        }
    }
}

// ---------------------------------------------------------------------------
// fp32 -> bf16 conversion
// ---------------------------------------------------------------------------
__global__ void f32_to_bf16_kernel(const float* __restrict__ in,
                                   bf16* __restrict__ out, int n)
{
    int i = blockIdx.x * 256 + threadIdx.x;
    if (i < n) out[i] = f2bf(in[i]);
}

// ---------------------------------------------------------------------------
// Embedding (return/state/action interleave) + embed LayerNorm
// ---------------------------------------------------------------------------
__global__ __launch_bounds__(256)
void embed_ln_kernel(const float* __restrict__ states,
                     const float* __restrict__ actions,
                     const float* __restrict__ rtg,
                     const float* __restrict__ erw, const float* __restrict__ erb,
                     const float* __restrict__ esw, const float* __restrict__ esb,
                     const float* __restrict__ eaw, const float* __restrict__ eab,
                     const float* __restrict__ lnw, const float* __restrict__ lnb,
                     float* __restrict__ x, bf16* __restrict__ xbf)
{
    const int row = blockIdx.x;        // 0..ROWS-1
    const int b   = row / L3;
    const int pos = row % L3;
    const int t   = pos / 3;
    const int typ = pos % 3;           // 0=return 1=state 2=action

    __shared__ float e[DM];
    __shared__ float sh[256];

    for (int d = threadIdx.x; d < DM; d += 256) {
        float v;
        if (typ == 0) {
            v = rtg[b * SEQ + t] * erw[d] + erb[d];
        } else if (typ == 1) {
            v = esb[d];
            const float* s = states + (size_t)(b * SEQ + t) * SDIM;
#pragma unroll
            for (int i = 0; i < SDIM; ++i) v += s[i] * esw[i * DM + d];
        } else {
            v = eab[d];
            const float* a = actions + (size_t)(b * SEQ + t) * ADIM;
#pragma unroll
            for (int i = 0; i < ADIM; ++i) v += a[i] * eaw[i * DM + d];
        }
        e[d] = v;
    }
    __syncthreads();

    sh[threadIdx.x] = e[threadIdx.x] + e[threadIdx.x + 256];
    __syncthreads();
    for (int st = 128; st > 0; st >>= 1) {
        if (threadIdx.x < st) sh[threadIdx.x] += sh[threadIdx.x + st];
        __syncthreads();
    }
    const float mu = sh[0] * (1.0f / DM);
    __syncthreads();
    const float d0 = e[threadIdx.x] - mu, d1 = e[threadIdx.x + 256] - mu;
    sh[threadIdx.x] = d0 * d0 + d1 * d1;
    __syncthreads();
    for (int st = 128; st > 0; st >>= 1) {
        if (threadIdx.x < st) sh[threadIdx.x] += sh[threadIdx.x + st];
        __syncthreads();
    }
    const float rinv = rsqrtf(sh[0] * (1.0f / DM) + 1e-5f);

    const size_t base = (size_t)row * DM;
    int d = threadIdx.x;
    float v0 = (e[d] - mu) * rinv * lnw[d] + lnb[d];
    x[base + d] = v0; xbf[base + d] = f2bf(v0);
    d += 256;
    float v1 = (e[d] - mu) * rinv * lnw[d] + lnb[d];
    x[base + d] = v1; xbf[base + d] = f2bf(v1);
}

// ---------------------------------------------------------------------------
// Final LayerNorm (ln_f)
// ---------------------------------------------------------------------------
__global__ __launch_bounds__(256)
void layernorm_kernel(const float* __restrict__ in,
                      const float* __restrict__ w, const float* __restrict__ b,
                      float* __restrict__ out)
{
    const int row = blockIdx.x;
    const size_t base = (size_t)row * DM;
    __shared__ float sh[256];

    const float e0 = in[base + threadIdx.x];
    const float e1 = in[base + threadIdx.x + 256];

    sh[threadIdx.x] = e0 + e1;
    __syncthreads();
    for (int st = 128; st > 0; st >>= 1) {
        if (threadIdx.x < st) sh[threadIdx.x] += sh[threadIdx.x + st];
        __syncthreads();
    }
    const float mu = sh[0] * (1.0f / DM);
    __syncthreads();
    const float d0 = e0 - mu, d1 = e1 - mu;
    sh[threadIdx.x] = d0 * d0 + d1 * d1;
    __syncthreads();
    for (int st = 128; st > 0; st >>= 1) {
        if (threadIdx.x < st) sh[threadIdx.x] += sh[threadIdx.x + st];
        __syncthreads();
    }
    const float rinv = rsqrtf(sh[0] * (1.0f / DM) + 1e-5f);

    out[base + threadIdx.x]       = d0 * rinv * w[threadIdx.x]       + b[threadIdx.x];
    out[base + threadIdx.x + 256] = d1 * rinv * w[threadIdx.x + 256] + b[threadIdx.x + 256];
}

// ---------------------------------------------------------------------------
// Causal depthwise conv (D_CONV=4) over stacked length L3 + SiLU.
// ---------------------------------------------------------------------------
__global__ void conv_silu_kernel(const float* __restrict__ xz,
                                 const float* __restrict__ cw,
                                 const float* __restrict__ cb,
                                 float* __restrict__ xc, bf16* __restrict__ xcbf)
{
    const int i = blockIdx.x * 256 + threadIdx.x;   // over ROWS*DI
    if (i >= ROWS * DI) return;
    const int d   = i & (DI - 1);
    const int row = i >> 10;
    const int l   = row % L3;

    float acc = cb[d];
#pragma unroll
    for (int k = 0; k < DC; ++k) {
        const int ls = l + k - (DC - 1);
        if (ls >= 0)
            acc += cw[d * DC + k] * xz[(size_t)(row + k - (DC - 1)) * (2 * DI) + d];
    }
    const float s = silu_f(acc);
    xc[i] = s; xcbf[i] = f2bf(s);
}

// ---------------------------------------------------------------------------
// Selective scan + skip (x*D) + gate (silu(z)) fused.
// ---------------------------------------------------------------------------
__global__ __launch_bounds__(256)
void scan_kernel(const float* __restrict__ dt,   // [ROWS, DI]
                 const float* __restrict__ dbl,  // [ROWS, 64]  (dt|B|C)
                 const float* __restrict__ xc,   // [ROWS, DI]
                 const float* __restrict__ xz,   // [ROWS, 2*DI] (z at +DI)
                 const float* __restrict__ A_log,// [DI, DS]
                 const float* __restrict__ Dp,   // [DI]
                 float* __restrict__ y, bf16* __restrict__ ybf)
{
    const int b = blockIdx.x;
    const int d = blockIdx.y * 256 + threadIdx.x;

    float a[DS], h[DS];
#pragma unroll
    for (int n = 0; n < DS; ++n) { a[n] = -__expf(A_log[d * DS + n]); h[n] = 0.0f; }
    const float Dv = Dp[d];

    __shared__ float Bs[DS], Cs[DS];

    for (int l = 0; l < L3; ++l) {
        const int row = b * L3 + l;
        if (threadIdx.x < DS)
            Bs[threadIdx.x] = dbl[(size_t)row * 64 + DTR + threadIdx.x];
        else if (threadIdx.x < 2 * DS)
            Cs[threadIdx.x - DS] = dbl[(size_t)row * 64 + DTR + DS + (threadIdx.x - DS)];
        __syncthreads();

        const float dtv = dt[(size_t)row * DI + d];
        const float xv  = xc[(size_t)row * DI + d];
        const float dx  = dtv * xv;
        float acc = 0.0f;
#pragma unroll
        for (int n = 0; n < DS; ++n) {
            h[n] = h[n] * __expf(dtv * a[n]) + dx * Bs[n];
            acc += h[n] * Cs[n];
        }
        const float zv = xz[(size_t)row * (2 * DI) + DI + d];
        const float v  = (acc + xv * Dv) * silu_f(zv);
        y[(size_t)row * DI + d]   = v;
        ybf[(size_t)row * DI + d] = f2bf(v);
        __syncthreads();
    }
}

// ---------------------------------------------------------------------------
// Head: pick state tokens (3t+1), project to 6 actions, tanh.
// ---------------------------------------------------------------------------
__global__ void predict_kernel(const float* __restrict__ xn,
                               const float* __restrict__ pw,
                               const float* __restrict__ pb,
                               float* __restrict__ out)
{
    const int i = blockIdx.x * 256 + threadIdx.x;   // BATCH*SEQ*ADIM
    if (i >= BATCH * SEQ * ADIM) return;
    const int j  = i % ADIM;
    const int bt = i / ADIM;
    const int b  = bt / SEQ, t = bt % SEQ;
    const size_t base = (size_t)(b * L3 + 3 * t + 1) * DM;
    float acc = pb[j];
    for (int d = 0; d < DM; ++d) acc += xn[base + d] * pw[d * ADIM + j];
    out[i] = tanhf(acc);
}

// ---------------------------------------------------------------------------
// Host launcher
// ---------------------------------------------------------------------------
extern "C" void kernel_launch(void* const* d_in, const int* in_sizes, int n_in,
                              void* d_out, int out_size, void* d_ws, size_t ws_size,
                              hipStream_t stream)
{
    (void)in_sizes; (void)n_in; (void)out_size; (void)ws_size;

    const float* states  = (const float*)d_in[0];
    const float* actions = (const float*)d_in[1];
    const float* rtg     = (const float*)d_in[2];
    const float* erw = (const float*)d_in[3];
    const float* erb = (const float*)d_in[4];
    const float* esw = (const float*)d_in[5];
    const float* esb = (const float*)d_in[6];
    const float* eaw = (const float*)d_in[7];
    const float* eab = (const float*)d_in[8];
    const float* elnw = (const float*)d_in[9];
    const float* elnb = (const float*)d_in[10];
    const float* lnfw = (const float*)d_in[11];
    const float* lnfb = (const float*)d_in[12];
    const float* paw  = (const float*)d_in[13];
    const float* pab  = (const float*)d_in[14];

    // -------- workspace carve-up --------
    char* wsp = (char*)d_ws;
    auto alloc = [&](size_t bytes) -> char* {
        char* p = wsp; wsp += (bytes + 255) & ~(size_t)255; return p;
    };
    float* x     = (float*)alloc((size_t)ROWS * DM * 4);        // residual stream
    bf16*  xbf   = (bf16* )alloc((size_t)ROWS * DM * 2);
    float* xz    = (float*)alloc((size_t)ROWS * 2 * DI * 4);
    float* xc    = (float*)alloc((size_t)ROWS * DI * 4);
    bf16*  xcbf  = (bf16* )alloc((size_t)ROWS * DI * 2);
    float* dbl   = (float*)alloc((size_t)ROWS * 64 * 4);
    bf16*  dblbf = (bf16* )alloc((size_t)ROWS * 64 * 2);
    float* dtb   = (float*)alloc((size_t)ROWS * DI * 4);
    float* yb    = (float*)alloc((size_t)ROWS * DI * 4);
    bf16*  ybf   = (bf16* )alloc((size_t)ROWS * DI * 2);
    float* xn    = (float*)alloc((size_t)ROWS * DM * 4);
    bf16*  w_in  = (bf16* )alloc((size_t)DM * 2 * DI * 2);      // packed fragments
    bf16*  w_xp  = (bf16* )alloc((size_t)DI * (DTR + 2 * DS) * 2);
    bf16*  w_dt  = (bf16* )alloc((size_t)DTR * DI * 2);
    bf16*  w_out = (bf16* )alloc((size_t)DI * DM * 2);

    const dim3 gblk(128);
    auto packB = [&](const float* src, bf16* dst, int K, int N) {
        const int frags = (K >> 5) * (N >> 4);
        pack_b_kernel<<<(frags + 3) / 4, gblk, 0, stream>>>(src, dst, K, N);
    };
    auto cvt = [&](const float* src, bf16* dst, int n) {
        f32_to_bf16_kernel<<<(n + 255) / 256, 256, 0, stream>>>(src, dst, n);
    };

    // -------- embedding + embed LN --------
    embed_ln_kernel<<<ROWS, 256, 0, stream>>>(states, actions, rtg,
                                              erw, erb, esw, esb, eaw, eab,
                                              elnw, elnb, x, xbf);

    // -------- layers --------
    for (int L = 0; L < NL; ++L) {
        const int base = 15 + 9 * L;
        const float* in_proj   = (const float*)d_in[base + 0];   // [512,2048]
        const float* conv_w    = (const float*)d_in[base + 1];   // [1024,4]
        const float* conv_b    = (const float*)d_in[base + 2];   // [1024]
        const float* x_proj    = (const float*)d_in[base + 3];   // [1024,64]
        const float* dt_proj_w = (const float*)d_in[base + 4];   // [32,1024]
        const float* dt_proj_b = (const float*)d_in[base + 5];   // [1024]
        const float* A_log     = (const float*)d_in[base + 6];   // [1024,16]
        const float* Dvec      = (const float*)d_in[base + 7];   // [1024]
        const float* out_proj  = (const float*)d_in[base + 8];   // [1024,512]

        packB(in_proj,   w_in,  DM,  2 * DI);
        packB(x_proj,    w_xp,  DI,  DTR + 2 * DS);
        packB(dt_proj_w, w_dt,  DTR, DI);
        packB(out_proj,  w_out, DI,  DM);

        // xz = x @ in_proj                       [12288,2048]
        gemm_wmma_bf16<2, 4, ACT_NONE, false, false, false>
            <<<dim3((2 * DI) / 256, ROWS / 32), gblk, 0, stream>>>(
                xbf, DM, w_in, nullptr, nullptr,
                xz, nullptr, ROWS, 2 * DI, DM);

        // xc = silu(causal depthwise conv(xs))   [12288,1024]
        conv_silu_kernel<<<(ROWS * DI) / 256, 256, 0, stream>>>(
            xz, conv_w, conv_b, xc, xcbf);

        // dbl = xc @ x_proj                      [12288,64]
        gemm_wmma_bf16<2, 1, ACT_NONE, false, false, false>
            <<<dim3(64 / 64, ROWS / 32), gblk, 0, stream>>>(
                xcbf, DI, w_xp, nullptr, nullptr,
                dbl, nullptr, ROWS, 64, DI);
        cvt(dbl, dblbf, ROWS * 64);

        // dt = softplus(dbl[:, :32] @ dt_proj_w + b)   [12288,1024]
        gemm_wmma_bf16<2, 4, ACT_SOFTPLUS, true, false, false>
            <<<dim3(DI / 256, ROWS / 32), gblk, 0, stream>>>(
                dblbf, 64, w_dt, dt_proj_b, nullptr,
                dtb, nullptr, ROWS, DI, DTR);

        // selective scan + skip + gate           [12288,1024]
        scan_kernel<<<dim3(BATCH, DI / 256), 256, 0, stream>>>(
            dtb, dbl, xc, xz, A_log, Dvec, yb, ybf);

        // x = y @ out_proj + x  (in-place residual)    [12288,512]
        gemm_wmma_bf16<2, 4, ACT_NONE, false, true, true>
            <<<dim3(DM / 256, ROWS / 32), gblk, 0, stream>>>(
                ybf, DI, w_out, nullptr, x,
                x, xbf, ROWS, DM, DI);
    }

    // -------- final LN + action head --------
    layernorm_kernel<<<ROWS, 256, 0, stream>>>(x, lnfw, lnfb, xn);
    predict_kernel<<<(BATCH * SEQ * ADIM + 255) / 256, 256, 0, stream>>>(
        xn, paw, pab, (float*)d_out);
}
